// BayesianSequenceModel_77506979823782
// MI455X (gfx1250) — matrix operand order repaired and checked by reading, not compiled, not run on hardware
//
#include <hip/hip_runtime.h>
#include <hip/hip_bf16.h>
#include <cstdint>

// ---------------------------------------------------------------------------
// Types for CDNA5 WMMA (wave32): 16x16x32 bf16 -> f32
// ---------------------------------------------------------------------------
typedef __attribute__((ext_vector_type(16))) __bf16 v16bf;
typedef __attribute__((ext_vector_type(8)))  float  v8f;
typedef __attribute__((__vector_size__(16))) int    i4x;   // int4 vector

struct Q32 { uint4 a, b; };   // 32 bytes == one v16bf fragment

#if __has_builtin(__builtin_amdgcn_global_load_async_to_lds_b128) && \
    __has_builtin(__builtin_amdgcn_s_wait_asynccnt)
#define USE_ASYNC_LDS 1
#else
#define USE_ASYNC_LDS 0
#endif

#if __has_builtin(__builtin_amdgcn_sched_barrier)
#define SCHED_FENCE() __builtin_amdgcn_sched_barrier(0)
#else
#define SCHED_FENCE()
#endif

__device__ __forceinline__ unsigned short f2bf(float f) {
  union { float f; unsigned int u; } v; v.f = f;
  unsigned int u = v.u;
  if ((u & 0x7F800000u) == 0x7F800000u && (u & 0x007FFFFFu))
    return (unsigned short)((u >> 16) | 0x0040u);           // quiet NaN
  unsigned int r = (u + 0x7FFFu + ((u >> 16) & 1u)) >> 16;  // RNE
  return (unsigned short)r;
}

__device__ __forceinline__ float sigmoidf(float x) { return 1.0f / (1.0f + expf(-x)); }

// ---------------------------------------------------------------------------
// Templated WMMA GEMM:  C[M x Npad] = A_bf16[M x K] * B_bf16[Npad x K]^T + bias
//   - per-wave tile 16 x (NT*16); 8 waves/block stacked along M (128 rows)
//   - all 8 waves share the same B tile -> B staged in LDS, double-buffered
//     via GLOBAL_LOAD_ASYNC_TO_LDS_B128 (ASYNCcnt) when available
//   - LDS B tile rows padded to 40 elems (80B) -> conflict-free ds_load_b128
//   - sched_barrier pins all NT fragment ds_loads before the WMMA chain so
//     a single s_wait_dscnt covers them and the NT WMMAs issue back-to-back
//   - K multiple of 32; lda/ldb multiples of 8 (16B-aligned fragment chunks)
//   - columns >= nmax masked on store -> direct strided output supported
// Fragment layouts per CDNA5 ISA 7.12.2:
//   A 16x32 bf16: lanes<16 hold K {0..7,16..23}, lanes>=16 K {8..15,24..31}
//   B 32x16 bf16: lane col N=L%16; lanes<16 K {0..7,16..23}, lanes>=16 K {8..15,24..31}
//   C f32: VGPR r, lane l -> M = r + 8*(l>=16), N = l%16
// ---------------------------------------------------------------------------
template <int NT>
__global__ __launch_bounds__(256) void wmma_gemm_bf16(
    const unsigned short* __restrict__ A, int lda,
    const unsigned short* __restrict__ B, int ldb,
    const float* __restrict__ bias,
    float* __restrict__ C, int ldc,
    int M, int K, int nmax)
{
  constexpr int ROWS = NT * 16;   // B tile rows (output columns covered)
  constexpr int RS   = 40;        // padded LDS row stride in elems (80 bytes)
  __shared__ __align__(16) unsigned short Bs[2][ROWS * RS];

  const int tid  = threadIdx.x;
  const int lane = tid & 31;
  const int wave = tid >> 5;
  const int m0 = (blockIdx.x * 8 + wave) * 16;
  const int n0 = blockIdx.y * ROWS;
  const int hv = lane >> 4;   // lane half
  const int lm = lane & 15;

  v8f acc[NT];
#pragma unroll
  for (int j = 0; j < NT; ++j) {
    const int n = n0 + j * 16 + lm;
    const float bv = (bias != nullptr && n < nmax) ? bias[n] : 0.0f;
#pragma unroll
    for (int e = 0; e < 8; ++e) acc[j][e] = bv;
  }

  const unsigned short* __restrict__ arow = A + (size_t)(m0 + lm) * lda;
  const int ntiles = K >> 5;

  // Issue one 16B chunk per thread: ROWS rows x 4 chunks of the B k-tile.
  auto issueB = [&](int buf, int kt) {
    if (tid < ROWS * 4) {
      const int row = tid >> 2, ch = tid & 3;
      const unsigned short* gp = B + (size_t)(n0 + row) * ldb + (kt << 5) + ch * 8;
      unsigned short* lp = &Bs[buf][row * RS + ch * 8];
#if USE_ASYNC_LDS
      __builtin_amdgcn_global_load_async_to_lds_b128(
          (__attribute__((address_space(1))) i4x*)(i4x*)(void*)const_cast<unsigned short*>(gp),
          (__attribute__((address_space(3))) i4x*)(i4x*)(void*)lp,
          0, 0);
#else
      *reinterpret_cast<uint4*>(lp) = *reinterpret_cast<const uint4*>(gp);
#endif
    }
  };

  issueB(0, 0);
  for (int kt = 0; kt < ntiles; ++kt) {
    const int  cur  = kt & 1;
    const bool more = (kt + 1 < ntiles);
    if (more) issueB(cur ^ 1, kt + 1);      // DMA next tile into other buffer
#if USE_ASYNC_LDS
    if (more) __builtin_amdgcn_s_wait_asynccnt(1);  // tile kt landed (in-order)
    else      __builtin_amdgcn_s_wait_asynccnt(0);
#endif
    __syncthreads();                        // tile kt visible to all 8 waves

    const int k0 = kt << 5;
    Q32 qa;
    qa.a = *reinterpret_cast<const uint4*>(arow + k0 + hv * 8);
    qa.b = *reinterpret_cast<const uint4*>(arow + k0 + 16 + hv * 8);
    const v16bf af = __builtin_bit_cast(v16bf, qa);
    if (more) __builtin_prefetch((const void*)(arow + k0 + 32), 0, 3);

    // Preload all NT B fragments into distinct registers, then fence the
    // scheduler so the ds_loads stay grouped ahead of the WMMA chain.
    v16bf bfrag[NT];
#pragma unroll
    for (int j = 0; j < NT; ++j) {
      const unsigned short* bp = &Bs[cur][(j * 16 + lm) * RS + hv * 8];
      Q32 qb;
      qb.a = *reinterpret_cast<const uint4*>(bp);
      qb.b = *reinterpret_cast<const uint4*>(bp + 16);
      bfrag[j] = __builtin_bit_cast(v16bf, qb);
    }
    SCHED_FENCE();
#pragma unroll
    for (int j = 0; j < NT; ++j) {
      acc[j] = __builtin_amdgcn_wmma_f32_16x16x32_bf16(
          false, af, false, bfrag[j], (short)0, acc[j], false, false);
    }
    SCHED_FENCE();
    __syncthreads();                        // reads done before buffer reuse
  }

#pragma unroll
  for (int j = 0; j < NT; ++j) {
    const int n = n0 + j * 16 + lm;
    if (n >= nmax) continue;
#pragma unroll
    for (int r = 0; r < 8; ++r) {
      const int m = m0 + r + hv * 8;
      C[(size_t)m * ldc + n] = acc[j][r];
    }
  }
}

// ---------------------------------------------------------------------------
// Per-column batch statistics (biased var, training-mode BN): one block/column
// stats layout: [mean(0..cols), rstd(0..cols)]
// ---------------------------------------------------------------------------
__global__ __launch_bounds__(256) void colstats_kernel(
    const float* __restrict__ X, int rows, int rowStride,
    float* __restrict__ stats, int cols)
{
  __shared__ float s1[256];
  __shared__ float s2[256];
  const int c = blockIdx.x;
  float sum = 0.f, sq = 0.f;
  for (int r = threadIdx.x; r < rows; r += 256) {
    const float v = X[(size_t)r * rowStride + c];
    sum += v; sq += v * v;
  }
  s1[threadIdx.x] = sum; s2[threadIdx.x] = sq;
  __syncthreads();
  for (int s = 128; s > 0; s >>= 1) {
    if (threadIdx.x < s) {
      s1[threadIdx.x] += s1[threadIdx.x + s];
      s2[threadIdx.x] += s2[threadIdx.x + s];
    }
    __syncthreads();
  }
  if (threadIdx.x == 0) {
    const float inv = 1.0f / (float)rows;
    const float m = s1[0] * inv;
    const float var = s2[0] * inv - m * m;
    stats[c] = m;
    stats[cols + c] = rsqrtf(var + 1e-5f);
  }
}

// BN apply (+optional ReLU) and pack to bf16 with zero K-padding up to `span`
__global__ __launch_bounds__(256) void bn_act_bf16_kernel(
    const float* __restrict__ X, int ldx, int cols,
    const float* __restrict__ stats,
    const float* __restrict__ g, const float* __restrict__ b,
    unsigned short* __restrict__ Y, int ldy, int span, int rows, int relu)
{
  const int total = rows * span;
  for (int idx = blockIdx.x * 256 + threadIdx.x; idx < total; idx += gridDim.x * 256) {
    const int r = idx / span, c = idx - r * span;
    float v = 0.f;
    if (c < cols) {
      const float x = X[(size_t)r * ldx + c];
      v = (x - stats[c]) * stats[cols + c] * g[c] + b[c];
      if (relu) v = fmaxf(v, 0.f);
    }
    Y[(size_t)r * ldy + c] = f2bf(v);
  }
}

// concat(a_t, z) -> fp32 scratch [N x 160]
__global__ __launch_bounds__(256) void concat_az_kernel(
    const float* __restrict__ A, int t, const float* __restrict__ z,
    float* __restrict__ X, int N, int T)
{
  const int total = N * 160;
  for (int idx = blockIdx.x * 256 + threadIdx.x; idx < total; idx += gridDim.x * 256) {
    const int n = idx / 160, c = idx - n * 160;
    float v = 0.f;
    if (c < 8)        v = A[((size_t)n * T + t) * 8 + c];
    else if (c < 136) v = z[(size_t)n * 128 + (c - 8)];
    X[idx] = v;
  }
}

// concat(x0, z_t) -> fp32 scratch [N x 160]
__global__ __launch_bounds__(256) void concat_xz_kernel(
    const float* __restrict__ x0, const float* __restrict__ z,
    float* __restrict__ X, int N)
{
  const int total = N * 160;
  for (int idx = blockIdx.x * 256 + threadIdx.x; idx < total; idx += gridDim.x * 256) {
    const int n = idx / 160, c = idx - n * 160;
    float v = 0.f;
    if (c < 21)       v = x0[(size_t)n * 21 + c];
    else if (c < 149) v = z[(size_t)n * 128 + (c - 21)];
    X[idx] = v;
  }
}

// LSTM elementwise: c <- sig(f)*c + sig(i)*tanh(g); h=sig(o)*tanh(c) packed to
// bf16 directly into the concatenated A operand (row stride 672, h section)
__global__ __launch_bounds__(256) void lstm_elem_kernel(
    const float* __restrict__ gates, float* __restrict__ c,
    unsigned short* __restrict__ hdst, int N, int H)
{
  const int total = N * H;
  for (int idx = blockIdx.x * 256 + threadIdx.x; idx < total; idx += gridDim.x * 256) {
    const int n = idx / H, k = idx - n * H;
    const float* gr = gates + (size_t)n * 4 * H;
    const float ig = sigmoidf(gr[k]);
    const float fg = sigmoidf(gr[H + k]);
    const float gg = tanhf(gr[2 * H + k]);
    const float og = sigmoidf(gr[3 * H + k]);
    const float cn = fg * c[idx] + ig * gg;
    c[idx] = cn;
    hdst[(size_t)n * 672 + k] = f2bf(og * tanhf(cn));
  }
}

// z <- z_loc + softplus(z_raw) * eps[:, t, :]
__global__ __launch_bounds__(256) void zupd_kernel(
    const float* __restrict__ zc, const float* __restrict__ eps, int t,
    float* __restrict__ z, int N, int T)
{
  const int total = N * 128;
  for (int idx = blockIdx.x * 256 + threadIdx.x; idx < total; idx += gridDim.x * 256) {
    const int n = idx / 128, k = idx - n * 128;
    const float loc = zc[(size_t)n * 256 + k];
    const float raw = zc[(size_t)n * 256 + 128 + k];
    const float sp = (raw > 20.f) ? raw : log1pf(expf(raw));
    z[idx] = loc + sp * eps[((size_t)n * T + t) * 128 + k];
  }
}

// fp32 weight [rows x cols] -> bf16 [outRows x ldy] at column offset, zero pad
__global__ __launch_bounds__(256) void pack_w_kernel(
    const float* __restrict__ W, int rows, int cols,
    unsigned short* __restrict__ Y, int ldy, int colOff, int span, int outRows)
{
  const int total = outRows * span;
  for (int idx = blockIdx.x * 256 + threadIdx.x; idx < total; idx += gridDim.x * 256) {
    const int r = idx / span, c = idx - r * span;
    const float v = (r < rows && c < cols) ? W[(size_t)r * cols + c] : 0.f;
    Y[(size_t)r * ldy + colOff + c] = f2bf(v);
  }
}

__global__ __launch_bounds__(256) void addvec_kernel(
    const float* __restrict__ a, const float* __restrict__ b,
    float* __restrict__ o, int n)
{
  for (int i = blockIdx.x * 256 + threadIdx.x; i < n; i += gridDim.x * 256)
    o[i] = a[i] + b[i];
}

// Broadcast init_z/init_h/init_c into state buffers (h in bf16 into A-concat)
__global__ __launch_bounds__(256) void init_state_kernel(
    const float* __restrict__ iz, const float* __restrict__ ih,
    const float* __restrict__ ic, float* __restrict__ z, float* __restrict__ c,
    unsigned short* __restrict__ hdst, int N)
{
  const int total = N * 512;
  for (int idx = blockIdx.x * 256 + threadIdx.x; idx < total; idx += gridDim.x * 256) {
    const int n = idx / 512, k = idx - n * 512;
    c[idx] = ic[k];
    hdst[(size_t)n * 672 + k] = f2bf(ih[k]);
    if (k < 128) z[(size_t)n * 128 + k] = iz[k];
  }
}

// ---------------------------------------------------------------------------
// Host orchestration
// ---------------------------------------------------------------------------
static inline size_t alignUp256(size_t x) { return (x + 255) & ~(size_t)255; }
static inline int nblk(int total) { return (total + 255) / 256; }

extern "C" void kernel_launch(void* const* d_in, const int* in_sizes, int n_in,
                              void* d_out, int out_size, void* d_ws, size_t ws_size,
                              hipStream_t stream) {
  (void)in_sizes; (void)n_in; (void)out_size; (void)ws_size;
  constexpr int N = 2048, T = 64, S = 21, H = 512;

  const float* x0     = (const float*)d_in[0];
  const float* A      = (const float*)d_in[1];
  const float* eps    = (const float*)d_in[2];
  const float* init_z = (const float*)d_in[3];
  const float* init_h = (const float*)d_in[4];
  const float* init_c = (const float*)d_in[5];
  const float* bn_in_g = (const float*)d_in[6];
  const float* bn_in_b = (const float*)d_in[7];
  const float* W_ih = (const float*)d_in[8];
  const float* W_hh = (const float*)d_in[9];
  const float* b_ih = (const float*)d_in[10];
  const float* b_hh = (const float*)d_in[11];
  const float* enc_w1 = (const float*)d_in[12];
  const float* enc_b1 = (const float*)d_in[13];
  const float* enc_w2 = (const float*)d_in[14];
  const float* enc_b2 = (const float*)d_in[15];
  const float* enc_wz = (const float*)d_in[16];
  const float* enc_bz = (const float*)d_in[17];
  const float* enc_bn2_g = (const float*)d_in[18];
  const float* enc_bn2_b = (const float*)d_in[19];
  const float* enc_bn3_g = (const float*)d_in[20];
  const float* enc_bn3_b = (const float*)d_in[21];
  const float* dec_w1 = (const float*)d_in[22];
  const float* dec_b1 = (const float*)d_in[23];
  const float* dec_w2 = (const float*)d_in[24];
  const float* dec_b2 = (const float*)d_in[25];
  const float* dec_w3 = (const float*)d_in[26];
  const float* dec_b3 = (const float*)d_in[27];
  const float* dec_w4 = (const float*)d_in[28];
  const float* dec_b4 = (const float*)d_in[29];
  const float* dec_w5 = (const float*)d_in[30];
  const float* dec_b5 = (const float*)d_in[31];
  const float* dec_bn1_g = (const float*)d_in[32];
  const float* dec_bn1_b = (const float*)d_in[33];
  const float* dec_bn2_g = (const float*)d_in[34];
  const float* dec_bn2_b = (const float*)d_in[35];
  const float* dec_bn3_g = (const float*)d_in[36];
  const float* dec_bn3_b = (const float*)d_in[37];
  const float* dec_bn4_g = (const float*)d_in[38];
  const float* dec_bn4_b = (const float*)d_in[39];
  const float* dec_bn5_g = (const float*)d_in[40];
  const float* dec_bn5_b = (const float*)d_in[41];

  float* out = (float*)d_out;

  // Workspace bump allocator
  char* base = (char*)d_ws;
  size_t off = 0;
  auto alloc = [&](size_t bytes) -> void* {
    void* p = base + off; off = alignUp256(off + bytes); return p;
  };
  typedef unsigned short u16;
  // bf16 weights (K padded to x32, rows padded to x16)
  u16* Wcat  = (u16*)alloc((size_t)2048 * 672 * 2);  // [W_ih(136->160) | W_hh(512)]
  u16* We1   = (u16*)alloc((size_t)256 * 512 * 2);
  u16* We2   = (u16*)alloc((size_t)128 * 256 * 2);
  u16* Wez   = (u16*)alloc((size_t)256 * 128 * 2);
  u16* Wd1   = (u16*)alloc((size_t)64 * 160 * 2);
  u16* Wd2   = (u16*)alloc((size_t)64 * 64 * 2);
  u16* Wd3   = (u16*)alloc((size_t)32 * 64 * 2);
  u16* Wd4   = (u16*)alloc((size_t)16 * 32 * 2);
  u16* Wd5   = (u16*)alloc((size_t)32 * 32 * 2);   // rows 21->32, cols 16->32
  float* bcomb = (float*)alloc(2048 * 4);
  // state
  float* zbuf = (float*)alloc((size_t)N * 128 * 4);
  float* cbuf = (float*)alloc((size_t)N * H * 4);
  u16*   Acat = (u16*)alloc((size_t)N * 672 * 2);  // [ri_bf16(160) | h_bf16(512)]
  // activations
  float* gates = (float*)alloc((size_t)N * 2048 * 4);
  float* xin   = (float*)alloc((size_t)N * 160 * 4);   // reused: lstm-in & dec-in
  u16*   xdb   = (u16*)alloc((size_t)N * 160 * 2);
  float* e1    = (float*)alloc((size_t)N * 256 * 4);
  u16*   e1b   = (u16*)alloc((size_t)N * 256 * 2);
  float* e2    = (float*)alloc((size_t)N * 128 * 4);
  u16*   e2b   = (u16*)alloc((size_t)N * 128 * 2);
  float* zc    = (float*)alloc((size_t)N * 256 * 4);
  float* y1    = (float*)alloc((size_t)N * 64 * 4);
  u16*   y1b   = (u16*)alloc((size_t)N * 64 * 2);
  float* y2    = (float*)alloc((size_t)N * 64 * 4);
  u16*   y2b   = (u16*)alloc((size_t)N * 64 * 2);
  float* y3    = (float*)alloc((size_t)N * 32 * 4);
  u16*   y3b   = (u16*)alloc((size_t)N * 32 * 2);
  float* y4    = (float*)alloc((size_t)N * 16 * 4);
  u16*   y4b   = (u16*)alloc((size_t)N * 32 * 2);
  float* stats = (float*)alloc(2 * 256 * 4);

  const dim3 blk(256);

  // ---- one-time per launch: pack weights to padded bf16 (deterministic) ----
  pack_w_kernel<<<nblk(2048 * 160), blk, 0, stream>>>(W_ih, 2048, 136, Wcat, 672, 0, 160, 2048);
  pack_w_kernel<<<nblk(2048 * 512), blk, 0, stream>>>(W_hh, 2048, 512, Wcat, 672, 160, 512, 2048);
  pack_w_kernel<<<nblk(256 * 512), blk, 0, stream>>>(enc_w1, 256, 512, We1, 512, 0, 512, 256);
  pack_w_kernel<<<nblk(128 * 256), blk, 0, stream>>>(enc_w2, 128, 256, We2, 256, 0, 256, 128);
  pack_w_kernel<<<nblk(256 * 128), blk, 0, stream>>>(enc_wz, 256, 128, Wez, 128, 0, 128, 256);
  pack_w_kernel<<<nblk(64 * 160), blk, 0, stream>>>(dec_w1, 64, 149, Wd1, 160, 0, 160, 64);
  pack_w_kernel<<<nblk(64 * 64), blk, 0, stream>>>(dec_w2, 64, 64, Wd2, 64, 0, 64, 64);
  pack_w_kernel<<<nblk(32 * 64), blk, 0, stream>>>(dec_w3, 32, 64, Wd3, 64, 0, 64, 32);
  pack_w_kernel<<<nblk(16 * 32), blk, 0, stream>>>(dec_w4, 16, 32, Wd4, 32, 0, 32, 16);
  pack_w_kernel<<<nblk(32 * 32), blk, 0, stream>>>(dec_w5, 21, 16, Wd5, 32, 0, 32, 32);
  addvec_kernel<<<nblk(2048), blk, 0, stream>>>(b_ih, b_hh, bcomb, 2048);
  init_state_kernel<<<nblk(N * 512), blk, 0, stream>>>(init_z, init_h, init_c,
                                                       zbuf, cbuf, Acat + 160, N);

  for (int t = 0; t < T; ++t) {
    // ---- guide step: BN(concat(a_t, z)) -> bf16 A operand (cols 0..159) ----
    concat_az_kernel<<<nblk(N * 160), blk, 0, stream>>>(A, t, zbuf, xin, N, T);
    colstats_kernel<<<136, blk, 0, stream>>>(xin, N, 160, stats, 136);
    bn_act_bf16_kernel<<<nblk(N * 160), blk, 0, stream>>>(
        xin, 160, 136, stats, bn_in_g, bn_in_b, Acat, 672, 160, N, 0);
    // ---- fused LSTM GEMM: gates = [ri|h] * [W_ih|W_hh]^T + (b_ih+b_hh) ----
    wmma_gemm_bf16<4><<<dim3(N / 128, 2048 / 64), blk, 0, stream>>>(
        Acat, 672, Wcat, 672, bcomb, gates, 2048, N, 672, 2048);
    lstm_elem_kernel<<<nblk(N * H), blk, 0, stream>>>(gates, cbuf, Acat + 160, N, H);
    // ---- encoder ----
    wmma_gemm_bf16<4><<<dim3(N / 128, 256 / 64), blk, 0, stream>>>(
        Acat + 160, 672, We1, 512, enc_b1, e1, 256, N, 512, 256);
    colstats_kernel<<<256, blk, 0, stream>>>(e1, N, 256, stats, 256);
    bn_act_bf16_kernel<<<nblk(N * 256), blk, 0, stream>>>(
        e1, 256, 256, stats, enc_bn2_g, enc_bn2_b, e1b, 256, 256, N, 1);
    wmma_gemm_bf16<4><<<dim3(N / 128, 128 / 64), blk, 0, stream>>>(
        e1b, 256, We2, 256, enc_b2, e2, 128, N, 256, 128);
    colstats_kernel<<<128, blk, 0, stream>>>(e2, N, 128, stats, 128);
    bn_act_bf16_kernel<<<nblk(N * 128), blk, 0, stream>>>(
        e2, 128, 128, stats, enc_bn3_g, enc_bn3_b, e2b, 128, 128, N, 1);
    wmma_gemm_bf16<4><<<dim3(N / 128, 256 / 64), blk, 0, stream>>>(
        e2b, 128, Wez, 128, enc_bz, zc, 256, N, 128, 256);
    zupd_kernel<<<nblk(N * 128), blk, 0, stream>>>(zc, eps, t, zbuf, N, T);
    // ---- decoder for this timestep (per-t batch stats) ----
    concat_xz_kernel<<<nblk(N * 160), blk, 0, stream>>>(x0, zbuf, xin, N);
    colstats_kernel<<<149, blk, 0, stream>>>(xin, N, 160, stats, 149);
    bn_act_bf16_kernel<<<nblk(N * 160), blk, 0, stream>>>(
        xin, 160, 149, stats, dec_bn1_g, dec_bn1_b, xdb, 160, 160, N, 0);
    wmma_gemm_bf16<4><<<dim3(N / 128, 1), blk, 0, stream>>>(
        xdb, 160, Wd1, 160, dec_b1, y1, 64, N, 160, 64);
    colstats_kernel<<<64, blk, 0, stream>>>(y1, N, 64, stats, 64);
    bn_act_bf16_kernel<<<nblk(N * 64), blk, 0, stream>>>(
        y1, 64, 64, stats, dec_bn2_g, dec_bn2_b, y1b, 64, 64, N, 1);
    wmma_gemm_bf16<4><<<dim3(N / 128, 1), blk, 0, stream>>>(
        y1b, 64, Wd2, 64, dec_b2, y2, 64, N, 64, 64);
    colstats_kernel<<<64, blk, 0, stream>>>(y2, N, 64, stats, 64);
    bn_act_bf16_kernel<<<nblk(N * 64), blk, 0, stream>>>(
        y2, 64, 64, stats, dec_bn3_g, dec_bn3_b, y2b, 64, 64, N, 1);
    wmma_gemm_bf16<2><<<dim3(N / 128, 1), blk, 0, stream>>>(
        y2b, 64, Wd3, 64, dec_b3, y3, 32, N, 64, 32);
    colstats_kernel<<<32, blk, 0, stream>>>(y3, N, 32, stats, 32);
    bn_act_bf16_kernel<<<nblk(N * 32), blk, 0, stream>>>(
        y3, 32, 32, stats, dec_bn4_g, dec_bn4_b, y3b, 32, 32, N, 1);
    wmma_gemm_bf16<1><<<dim3(N / 128, 1), blk, 0, stream>>>(
        y3b, 32, Wd4, 32, dec_b4, y4, 16, N, 32, 16);
    colstats_kernel<<<16, blk, 0, stream>>>(y4, N, 16, stats, 16);
    bn_act_bf16_kernel<<<nblk(N * 32), blk, 0, stream>>>(
        y4, 16, 16, stats, dec_bn5_g, dec_bn5_b, y4b, 32, 32, N, 1);
    // final layer writes directly into strided d_out[:, t, 0:21]
    wmma_gemm_bf16<2><<<dim3(N / 128, 1), blk, 0, stream>>>(
        y4b, 32, Wd5, 32, dec_b5, out + (size_t)t * S, T * S, N, 32, S);
  }
}
